// GraphBlock_16054587752807
// MI455X (gfx1250) — matrix operand order repaired and checked by reading, not compiled, not run on hardware
//
#include <hip/hip_runtime.h>

// ---------------------------------------------------------------------------
// Problem constants
// ---------------------------------------------------------------------------
#define BB   64
#define TT   196
#define JJ   22
#define DD   256
#define HH   8
#define HD   32
#define DFF  1024
#define JP   32                   // joints padded to 32 (2 x 16 M-tiles)
#define NFRM (BB*TT)              // 12544
#define MTOT (BB*TT*JJ)           // 275968 rows (divisible by 64)
#define RT   64                   // FFN rows per block (4 x 16 M-tiles)

typedef __attribute__((ext_vector_type(16))) __bf16 v16bf;
typedef __attribute__((ext_vector_type(8)))  float  v8f;

// ---------------------------------------------------------------------------
// Helpers
// ---------------------------------------------------------------------------
__device__ __forceinline__ unsigned short f2bf(float f) {
    unsigned int u = __float_as_uint(f);
    u += 0x7FFFu + ((u >> 16) & 1u);      // round-to-nearest-even
    return (unsigned short)(u >> 16);
}

__device__ __forceinline__ v8f zero8() {
    v8f z = {0.f,0.f,0.f,0.f,0.f,0.f,0.f,0.f};
    return z;
}

union BFrag { v16bf v; unsigned int u[8]; };

// A-matrix fragment 16x32 bf16 (M x K), rows striped lanes 0..15, k = hi*16 + s (contiguous, matches load_b_frag)
// element source: src[(m0+lo)*ld + k0 + k]
__device__ __forceinline__ v16bf load_a_frag(const unsigned short* src, int ld,
                                             int m0, int k0, int lo, int hi) {
    BFrag r;
    const unsigned short* row = src + (m0 + lo) * ld + k0;
#pragma unroll
    for (int j = 0; j < 8; ++j)
        r.u[j]     = *(const unsigned int*)(row + hi*16 + 2*j);
    return r.v;
}

// B-matrix fragment 32x16 bf16 (K x N), B(k,n) = src[(n0+lo)*ld + k0 + k]
// (weights stored row-major [N_out, K_in] => pairs (k,k+1) contiguous)
__device__ __forceinline__ v16bf load_b_frag(const unsigned short* src, int ld,
                                             int n0, int k0, int lo, int hi) {
    BFrag r;
    const unsigned short* row = src + (n0 + lo) * ld + k0 + hi*16;
#pragma unroll
    for (int v = 0; v < 8; ++v)
        r.u[v] = *(const unsigned int*)(row + 2*v);
    return r.v;
}

__device__ __forceinline__ v8f wmma_bf16(v16bf a, v16bf b, v8f c) {
    return __builtin_amdgcn_wmma_f32_16x16x32_bf16(false, a, false, b,
                                                   (short)0, c, false, false);
}

// ---------------------------------------------------------------------------
// Kernel 0: cast all weights fp32 -> bf16 into workspace
// layout: wqkv[768*256] | wout[256*256] | ffw1[1024*256] | ffw2[256*1024]
// ---------------------------------------------------------------------------
#define OFF_WQKV 0
#define OFF_WOUT 196608
#define OFF_W1   262144
#define OFF_W2   524288
#define W_TOTAL  786432

__global__ __launch_bounds__(256) void cast_weights_kernel(
    const float* __restrict__ wq, const float* __restrict__ wo,
    const float* __restrict__ w1, const float* __restrict__ w2,
    unsigned short* __restrict__ dst)
{
    int i = blockIdx.x * 256 + threadIdx.x;
    if (i >= W_TOTAL) return;
    float v;
    if      (i < OFF_WOUT) v = wq[i - OFF_WQKV];
    else if (i < OFF_W1)   v = wo[i - OFF_WOUT];
    else if (i < OFF_W2)   v = w1[i - OFF_W1];
    else                   v = w2[i - OFF_W2];
    dst[i] = f2bf(v);
}

// ---------------------------------------------------------------------------
// Kernel 1: per-frame fused  LN1 -> QKV -> masked MHA -> out_proj -> +residual
// one workgroup (8 waves) per frame; wave w owns head w in attention stage.
// LDS (96 KB total, 3 workgroups/WGP):
//   [0,32K)   : hS [32][256] bf16  (dead after QKV)   \  aliased by
//               yS [32][256] bf16  (live from C3)     /  scS [8][32][32] f32
//   [32K,80K) : qkvS [32][768] bf16
//   [80K,96K) : pS [8][32][32] bf16
// ---------------------------------------------------------------------------
__global__ __launch_bounds__(256) void attn_kernel(
    const float* __restrict__ x, const int* __restrict__ adj,
    const unsigned short* __restrict__ wqkv, const float* __restrict__ bqkv,
    const unsigned short* __restrict__ wout, const float* __restrict__ bout,
    const float* __restrict__ ln1g, const float* __restrict__ ln1b,
    float* __restrict__ out)
{
    extern __shared__ char smem_raw[];
    unsigned short* hS   = (unsigned short*)smem_raw;        // [32][256] 16 KB
    unsigned short* yS   = hS + JP*DD;                       // [32][256] 16 KB
    float*          scS  = (float*)smem_raw;                 // [8][32][32] 32 KB (alias)
    unsigned short* qkvS = (unsigned short*)(smem_raw + 32768);  // [32][768] 48 KB
    unsigned short* pS   = qkvS + JP*3*DD;                   // [8][32][32] 16 KB

    const int f = blockIdx.x;
    const float* xf = x   + (size_t)f * JJ * DD;
    float*       of = out + (size_t)f * JJ * DD;

    const int tid  = threadIdx.x;
    const int wave = tid >> 5;
    const int lane = tid & 31;
    const int lo   = lane & 15, hi = lane >> 4;

    // ---- Stage A: LN1 + cast to bf16 (pad rows 22..31 with zeros) ----
    for (int r = wave; r < JP; r += 8) {
        if (r < JJ) {
            float v[8]; float s = 0.f, ss = 0.f;
#pragma unroll
            for (int i = 0; i < 8; ++i) {
                v[i] = xf[r*DD + lane + 32*i];
                s += v[i]; ss += v[i]*v[i];
            }
#pragma unroll
            for (int o = 16; o > 0; o >>= 1) {
                s  += __shfl_xor(s,  o, 32);
                ss += __shfl_xor(ss, o, 32);
            }
            float mean = s * (1.f/256.f);
            float var  = ss * (1.f/256.f) - mean*mean;
            float rstd = rsqrtf(var + 1e-5f);
#pragma unroll
            for (int i = 0; i < 8; ++i) {
                int c = lane + 32*i;
                hS[r*DD + c] = f2bf((v[i]-mean)*rstd*ln1g[c] + ln1b[c]);
            }
        } else {
#pragma unroll
            for (int i = 0; i < 8; ++i) hS[r*DD + lane + 32*i] = 0;
        }
    }
    __syncthreads();

    // ---- Stage B: QKV = h[32,256] @ Wqkv^T[256,768] + bias -> LDS bf16 ----
    for (int nt = wave*6; nt < wave*6 + 6; ++nt) {
        int n0 = nt * 16;
        v8f acc0 = zero8(), acc1 = zero8();
#pragma unroll
        for (int k0 = 0; k0 < DD; k0 += 32) {
            v16bf a0 = load_a_frag(hS, DD, 0,  k0, lo, hi);
            v16bf a1 = load_a_frag(hS, DD, 16, k0, lo, hi);
            v16bf bb = load_b_frag(wqkv, DD, n0, k0, lo, hi);
            acc0 = wmma_bf16(a0, bb, acc0);
            acc1 = wmma_bf16(a1, bb, acc1);
        }
        float bias = bqkv[n0 + lo];
#pragma unroll
        for (int v = 0; v < 8; ++v) {
            int r = v + hi*8;
            qkvS[(r     )*3*DD + n0 + lo] = f2bf(acc0[v] + bias);
            qkvS[(r + 16)*3*DD + n0 + lo] = f2bf(acc1[v] + bias);
        }
    }
    __syncthreads();      // hS dead from here; scS may now live in its place

    // ---- Stage C1: scores = Q @ K^T / sqrt(HD), head = wave ----
    {
        const int h = wave;
        const float scale = 0.17677669529663687f;   // 1/sqrt(32)
        v16bf a0 = load_a_frag(qkvS, 3*DD, 0,  h*HD, lo, hi);        // Q rows
        v16bf a1 = load_a_frag(qkvS, 3*DD, 16, h*HD, lo, hi);
        v16bf b0 = load_b_frag(qkvS, 3*DD, 0,  DD + h*HD, lo, hi);   // K rows as cols
        v16bf b1 = load_b_frag(qkvS, 3*DD, 16, DD + h*HD, lo, hi);
        v8f s00 = wmma_bf16(a0, b0, zero8());
        v8f s01 = wmma_bf16(a0, b1, zero8());
        v8f s10 = wmma_bf16(a1, b0, zero8());
        v8f s11 = wmma_bf16(a1, b1, zero8());
        float* sch = scS + h*JP*JP;
#pragma unroll
        for (int v = 0; v < 8; ++v) {
            int r = v + hi*8;
            sch[(r     )*JP + lo     ] = s00[v] * scale;
            sch[(r     )*JP + 16 + lo] = s01[v] * scale;
            sch[(r + 16)*JP + lo     ] = s10[v] * scale;
            sch[(r + 16)*JP + 16 + lo] = s11[v] * scale;
        }
    }
    __syncthreads();

    // ---- Stage C2: masked softmax (head = wave, row = lane) ----
    {
        const int h = wave, r = lane;
        const float* sch = scS + h*JP*JP;
        unsigned short* ph = pS + h*JP*JP;
        if (r < JJ) {
            float vals[JJ];
            float mx = -1e30f;
#pragma unroll
            for (int c = 0; c < JJ; ++c) {
                float sv = (adj[r*JJ + c] > 0) ? sch[r*JP + c] : -1e30f;
                vals[c] = sv;
                mx = fmaxf(mx, sv);
            }
            float sum = 0.f;
#pragma unroll
            for (int c = 0; c < JJ; ++c) {
                float e = (vals[c] <= -1e29f) ? 0.f : __expf(vals[c] - mx);
                vals[c] = e; sum += e;
            }
            float inv = 1.f / sum;
#pragma unroll
            for (int c = 0; c < JJ; ++c) ph[r*JP + c] = f2bf(vals[c] * inv);
#pragma unroll
            for (int c = JJ; c < JP; ++c) ph[r*JP + c] = 0;
        } else {
#pragma unroll
            for (int c = 0; c < JP; ++c) ph[r*JP + c] = 0;
        }
    }
    __syncthreads();      // scS dead from here; yS may now live in its place

    // ---- Stage C3: y = probs @ V  (K = 32 joints, N = 32 head-dims) ----
    {
        const int h = wave;
        const unsigned short* ph = pS + h*JP*JP;
        v16bf a0 = load_a_frag(ph, JP, 0,  0, lo, hi);
        v16bf a1 = load_a_frag(ph, JP, 16, 0, lo, hi);
#pragma unroll
        for (int nt = 0; nt < 2; ++nt) {
            int n = nt*16 + lo;                       // head-dim column
            BFrag bu;
#pragma unroll
            for (int v = 0; v < 8; ++v) {             // V strided: pack u16 pairs
                int k = hi*16 + 2*v;                  // joint index
                unsigned e0 = qkvS[(k    )*3*DD + 2*DD + h*HD + n];
                unsigned e1 = qkvS[(k + 1)*3*DD + 2*DD + h*HD + n];
                bu.u[v] = e0 | (e1 << 16);
            }
            v8f y0 = wmma_bf16(a0, bu.v, zero8());
            v8f y1 = wmma_bf16(a1, bu.v, zero8());
#pragma unroll
            for (int v = 0; v < 8; ++v) {
                int r = v + hi*8;
                yS[(r     )*DD + h*HD + nt*16 + lo] = f2bf(y0[v]);
                yS[(r + 16)*DD + h*HD + nt*16 + lo] = f2bf(y1[v]);
            }
        }
    }
    __syncthreads();

    // ---- Stage D: out = y @ Wout^T + bias + x (residual), rows < 22 ----
    for (int nt = wave*2; nt < wave*2 + 2; ++nt) {
        int n0 = nt * 16;
        v8f acc0 = zero8(), acc1 = zero8();
#pragma unroll
        for (int k0 = 0; k0 < DD; k0 += 32) {
            v16bf a0 = load_a_frag(yS, DD, 0,  k0, lo, hi);
            v16bf a1 = load_a_frag(yS, DD, 16, k0, lo, hi);
            v16bf bb = load_b_frag(wout, DD, n0, k0, lo, hi);
            acc0 = wmma_bf16(a0, bb, acc0);
            acc1 = wmma_bf16(a1, bb, acc1);
        }
        float bias = bout[n0 + lo];
#pragma unroll
        for (int v = 0; v < 8; ++v) {
            int r = v + hi*8;                         // 0..15, always < 22
            of[r*DD + n0 + lo] = acc0[v] + bias + xf[r*DD + n0 + lo];
            int r2 = 16 + v + hi*8;
            if (r2 < JJ)
                of[r2*DD + n0 + lo] = acc1[v] + bias + xf[r2*DD + n0 + lo];
        }
    }
}

// ---------------------------------------------------------------------------
// Kernel 2: fused FFN over 64-row tiles:  x += W2 @ gelu(W1 @ LN2(x))
// intermediate [64,1024] lives entirely in LDS (bf16); each B-fragment load
// feeds 4 WMMAs (M = 4 tiles). Residual re-read from global (own rows only).
// LDS: h2 32 KB + g 128 KB = 160 KB  -> 2 workgroups/WGP
// ---------------------------------------------------------------------------
__global__ __launch_bounds__(256) void ffn_kernel(
    const unsigned short* __restrict__ w1, const float* __restrict__ b1,
    const unsigned short* __restrict__ w2, const float* __restrict__ b2,
    const float* __restrict__ ln2g, const float* __restrict__ ln2b,
    float* __restrict__ out)
{
    extern __shared__ char smem_raw[];
    unsigned short* h2 = (unsigned short*)smem_raw;   // [64][256]  bf16 32 KB
    unsigned short* g  = h2 + RT*DD;                  // [64][1024] bf16 128 KB

    float* xb = out + (size_t)blockIdx.x * RT * DD;

    const int tid  = threadIdx.x;
    const int wave = tid >> 5;
    const int lane = tid & 31;
    const int lo   = lane & 15, hi = lane >> 4;

    // ---- LN2 + cast to bf16 ----
    for (int r = wave; r < RT; r += 8) {
        float v[8]; float s = 0.f, ss = 0.f;
#pragma unroll
        for (int i = 0; i < 8; ++i) {
            v[i] = xb[r*DD + lane + 32*i];
            s += v[i]; ss += v[i]*v[i];
        }
#pragma unroll
        for (int o = 16; o > 0; o >>= 1) {
            s  += __shfl_xor(s,  o, 32);
            ss += __shfl_xor(ss, o, 32);
        }
        float mean = s * (1.f/256.f);
        float var  = ss * (1.f/256.f) - mean*mean;
        float rstd = rsqrtf(var + 1e-5f);
#pragma unroll
        for (int i = 0; i < 8; ++i) {
            int c = lane + 32*i;
            h2[r*DD + c] = f2bf((v[i]-mean)*rstd*ln2g[c] + ln2b[c]);
        }
    }
    __syncthreads();

    // ---- GEMM1 + exact GELU -> g ----
    for (int nt = wave*8; nt < wave*8 + 8; ++nt) {
        int n0 = nt * 16;
        v8f acc[4];
#pragma unroll
        for (int mt = 0; mt < 4; ++mt) acc[mt] = zero8();
#pragma unroll
        for (int k0 = 0; k0 < DD; k0 += 32) {
            v16bf bb = load_b_frag(w1, DD, n0, k0, lo, hi);
#pragma unroll
            for (int mt = 0; mt < 4; ++mt) {
                v16bf a = load_a_frag(h2, DD, mt*16, k0, lo, hi);
                acc[mt] = wmma_bf16(a, bb, acc[mt]);
            }
        }
        float bias = b1[n0 + lo];
#pragma unroll
        for (int mt = 0; mt < 4; ++mt) {
#pragma unroll
            for (int v = 0; v < 8; ++v) {
                float u = acc[mt][v] + bias;
                u = 0.5f * u * (1.f + erff(u * 0.70710678118654752f));
                int r = mt*16 + v + hi*8;
                g[r*DFF + n0 + lo] = f2bf(u);
            }
        }
    }
    __syncthreads();

    // ---- GEMM2 + residual (in place, own rows only) ----
    for (int nt = wave*2; nt < wave*2 + 2; ++nt) {
        int n0 = nt * 16;
        v8f acc[4];
#pragma unroll
        for (int mt = 0; mt < 4; ++mt) acc[mt] = zero8();
#pragma unroll 4
        for (int k0 = 0; k0 < DFF; k0 += 32) {
            v16bf bb = load_b_frag(w2, DFF, n0, k0, lo, hi);
#pragma unroll
            for (int mt = 0; mt < 4; ++mt) {
                v16bf a = load_a_frag(g, DFF, mt*16, k0, lo, hi);
                acc[mt] = wmma_bf16(a, bb, acc[mt]);
            }
        }
        float bias = b2[n0 + lo];
#pragma unroll
        for (int mt = 0; mt < 4; ++mt) {
#pragma unroll
            for (int v = 0; v < 8; ++v) {
                int idx = (mt*16 + v + hi*8)*DD + n0 + lo;
                xb[idx] = acc[mt][v] + bias + xb[idx];
            }
        }
    }
}

// ---------------------------------------------------------------------------
// Launch
// ---------------------------------------------------------------------------
extern "C" void kernel_launch(void* const* d_in, const int* in_sizes, int n_in,
                              void* d_out, int out_size, void* d_ws, size_t ws_size,
                              hipStream_t stream) {
    const float* x     = (const float*)d_in[0];
    const int*   adj   = (const int*)  d_in[1];
    const float* in_w  = (const float*)d_in[2];
    const float* in_b  = (const float*)d_in[3];
    const float* out_w = (const float*)d_in[4];
    const float* out_b = (const float*)d_in[5];
    const float* ln1g  = (const float*)d_in[6];
    const float* ln1b  = (const float*)d_in[7];
    const float* ln2g  = (const float*)d_in[8];
    const float* ln2b  = (const float*)d_in[9];
    const float* fw1   = (const float*)d_in[10];
    const float* fb1   = (const float*)d_in[11];
    const float* fw2   = (const float*)d_in[12];
    const float* fb2   = (const float*)d_in[13];
    float* out = (float*)d_out;

    unsigned short* wbf = (unsigned short*)d_ws;
    unsigned short* wq  = wbf + OFF_WQKV;
    unsigned short* wo  = wbf + OFF_WOUT;
    unsigned short* w1  = wbf + OFF_W1;
    unsigned short* w2  = wbf + OFF_W2;

    cast_weights_kernel<<<(W_TOTAL + 255)/256, 256, 0, stream>>>(in_w, out_w, fw1, fw2, wbf);

    // 96 KB dynamic LDS: (hS|yS|scores alias 32K) + qkv(48K) + probs(16K)
    attn_kernel<<<NFRM, 256, 98304, stream>>>(x, adj, wq, in_b, wo, out_b,
                                              ln1g, ln1b, out);

    // 160 KB dynamic LDS: h2(32K) + g(128K)
    ffn_kernel<<<MTOT/RT, 256, 163840, stream>>>(w1, fb1, w2, fb2,
                                                 ln2g, ln2b, out);
}
